// DLRMPrefetcher_86930138071567
// MI455X (gfx1250) — matrix operand order repaired and checked by reading, not compiled, not run on hardware
//
#include <hip/hip_runtime.h>
#include <hip/hip_bf16.h>

// ---------------- CDNA5 WMMA plumbing ----------------
typedef float  v8f   __attribute__((ext_vector_type(8)));
typedef __bf16 v16bf __attribute__((ext_vector_type(16)));

#define DEV static __device__ __forceinline__

#if defined(__has_builtin)
#if __has_builtin(__builtin_amdgcn_global_load_async_to_lds_b128) && \
    __has_builtin(__builtin_amdgcn_s_wait_asynccnt)
#define USE_ASYNC 1
#endif
#endif
#ifndef USE_ASYNC
#define USE_ASYNC 0
#endif

// Pointer types the async builtin expects: int4* in global (as1) / LDS (as3).
typedef int v4i_vs __attribute__((vector_size(16)));
typedef __attribute__((address_space(1))) v4i_vs gv4i;   // global int4
typedef __attribute__((address_space(3))) v4i_vs lv4i;   // LDS int4

// global(bf16,16B-aligned) -> LDS, 16 bytes per lane. Async path uses the
// gfx1250 ASYNCcnt-tracked direct-to-LDS copy (no VGPR round trip).
DEV void cp_b128(const __bf16* g, __bf16* l) {
#if USE_ASYNC
  __builtin_amdgcn_global_load_async_to_lds_b128((gv4i*)g, (lv4i*)l, 0, 0);
#else
  *(uint4*)l = *(const uint4*)g;
#endif
}
DEV void cp_wait() {
#if USE_ASYNC
  __builtin_amdgcn_s_wait_asynccnt(0);
#endif
}

DEV v8f zero8() { v8f z = {0.f,0.f,0.f,0.f,0.f,0.f,0.f,0.f}; return z; }

DEV v8f wmma_bf16(v16bf a, v16bf b, v8f c) {
  return __builtin_amdgcn_wmma_f32_16x16x32_bf16(false, a, false, b, (short)0, c,
                                                 false, false);
}

// A fragment: 16x32 (MxK) bf16, row-major in LDS with leading dim ld.
DEV v16bf load_a(const __bf16* base, int ld) {
  int l = threadIdx.x & 31;
  const __bf16* rp = base + (l & 15) * ld;
  int kb = (l < 16) ? 0 : 16;
  v16bf a;
#pragma unroll
  for (int j = 0; j < 4; ++j) { int k = kb + 2*j;          a[2*j] = rp[k]; a[2*j+1] = rp[k+1]; }
#pragma unroll
  for (int j = 4; j < 8; ++j) { int k = kb + 2*j;          a[2*j] = rp[k]; a[2*j+1] = rp[k+1]; }
  return a;
}

DEV v16bf load_a_clamp(const __bf16* base, int ld, int kmax) {
  int l = threadIdx.x & 31;
  const __bf16* rp = base + (l & 15) * ld;
  int kb = (l < 16) ? 0 : 16;
  v16bf a;
#pragma unroll
  for (int j = 0; j < 4; ++j) {
    int k = kb + 2*j;
    a[2*j]   = (k   < kmax) ? rp[k]   : (__bf16)0.0f;
    a[2*j+1] = (k+1 < kmax) ? rp[k+1] : (__bf16)0.0f;
  }
#pragma unroll
  for (int j = 4; j < 8; ++j) {
    int k = kb + 2*j;
    a[2*j]   = (k   < kmax) ? rp[k]   : (__bf16)0.0f;
    a[2*j+1] = (k+1 < kmax) ? rp[k+1] : (__bf16)0.0f;
  }
  return a;
}

// B fragment from LDS tile stored [n][k]: B[kk][n] = base[n*ld + kk]
DEV v16bf load_b_nk(const __bf16* base, int ld, int kmax) {
  int l = threadIdx.x & 31;
  const __bf16* rp = base + (l & 15) * ld;
  int kb = (l < 16) ? 0 : 16;
  v16bf b;
#pragma unroll
  for (int j = 0; j < 8; ++j) {
    int k = kb + 2*j;
    b[2*j]   = (k   < kmax) ? rp[k]   : (__bf16)0.0f;
    b[2*j+1] = (k+1 < kmax) ? rp[k+1] : (__bf16)0.0f;
  }
  return b;
}

// B fragment from LDS tile stored [k][n]: B[kk][n] = base[kk*ld + n]
DEV v16bf load_b_kn(const __bf16* base, int ld, int kmax) {
  int l = threadIdx.x & 31;
  int n = l & 15;
  int kb = (l < 16) ? 0 : 16;
  v16bf b;
#pragma unroll
  for (int j = 0; j < 8; ++j) {
    int k = kb + 2*j;
    b[2*j]   = (k   < kmax) ? base[k*ld + n]     : (__bf16)0.0f;
    b[2*j+1] = (k+1 < kmax) ? base[(k+1)*ld + n] : (__bf16)0.0f;
  }
  return b;
}

// ---------------- model constants ----------------
#define SS   200
#define SP   208          // padded seq (13 * 16)
#define DDIM 128
#define MRows 25600       // B*S

// ---------------- kernels ----------------

// f32 -> bf16 weight mirror (runs once per launch; ~2.5M elements total)
__global__ __launch_bounds__(256) void k_cvt(const float* __restrict__ s,
                                             __bf16* __restrict__ d, int n)
{
  int i = blockIdx.x * 256 + threadIdx.x;
  int stride = gridDim.x * 256;
  for (; i < n; i += stride) d[i] = (__bf16)s[i];
}

// Gather embeddings, produce f32 + bf16 activations.
__global__ __launch_bounds__(256) void k_gather(
    const int* __restrict__ tseq, const int* __restrict__ iseq,
    const float* __restrict__ te, const float* __restrict__ ie,
    float* __restrict__ xt, float* __restrict__ xi,
    __bf16* __restrict__ xtb, __bf16* __restrict__ xib)
{
  size_t r = blockIdx.x;
  int t = threadIdx.x, d = t & 127;
  if (t < 128) {
    float v = te[(size_t)tseq[r] * 128 + d];
    xt[r*128 + d] = v; xtb[r*128 + d] = (__bf16)v;
  } else {
    float v = ie[(size_t)iseq[r] * 128 + d];
    xi[r*128 + d] = v; xib[r*128 + d] = (__bf16)v;
  }
}

// WMMA GEMM: C[M,N] = A[M,K](bf16) @ W[N,K](bf16)^T + bias, opt relu/bf16 copy.
// 64x64 tile per 256-thread block; A/W staged via async global->LDS b128.
__global__ __launch_bounds__(256) void k_gemm(
    const __bf16* __restrict__ A, const __bf16* __restrict__ W,
    const float* __restrict__ bias, float* __restrict__ C,
    __bf16* __restrict__ Cb, int M, int N, int K, int relu)
{
  __shared__ __align__(16) __bf16 Al[64*32];
  __shared__ __align__(16) __bf16 Wl[64*32];
  int tid = threadIdx.x;
  int w = tid >> 5, l = tid & 31;
  int m0 = blockIdx.y * 64, n0 = blockIdx.x * 64;
  int mt = w >> 1, np = w & 1;
  v8f acc0 = zero8(), acc1 = zero8();
  int row = tid >> 2, kq = (tid & 3) * 8;
  for (int k0 = 0; k0 < K; k0 += 32) {
    cp_b128(A + (size_t)(m0 + row) * K + k0 + kq, &Al[row*32 + kq]);
    cp_b128(W + (size_t)(n0 + row) * K + k0 + kq, &Wl[row*32 + kq]);
    cp_wait();
    __syncthreads();
    v16bf af = load_a(&Al[mt*16*32], 32);
    v16bf b0 = load_b_nk(&Wl[(np*32)*32], 32, 32);
    v16bf b1 = load_b_nk(&Wl[(np*32+16)*32], 32, 32);
    acc0 = wmma_bf16(af, b0, acc0);
    acc1 = wmma_bf16(af, b1, acc1);
    __syncthreads();
  }
  int rowoff = (l < 16) ? 0 : 8;
  int c0 = n0 + np*32 + (l & 15);
#pragma unroll
  for (int r = 0; r < 8; ++r) {
    int m = m0 + mt*16 + rowoff + r;
    float v0 = acc0[r] + bias[c0];
    float v1 = acc1[r] + bias[c0 + 16];
    if (relu) { v0 = fmaxf(v0, 0.f); v1 = fmaxf(v1, 0.f); }
    C[(size_t)m*N + c0]      = v0;
    C[(size_t)m*N + c0 + 16] = v1;
    if (Cb) { Cb[(size_t)m*N + c0]      = (__bf16)v0;
              Cb[(size_t)m*N + c0 + 16] = (__bf16)v1; }
  }
}

// Attention, one block per (b,h). K/V/scores LDS-resident, scores & PV via WMMA.
__global__ __launch_bounds__(256) void k_attn(const float* __restrict__ qkv,
                                              __bf16* __restrict__ Ob)
{
  __shared__ __align__(16) __bf16 Kl[SP*16];
  __shared__ __align__(16) __bf16 Vl[SP*16];
  __shared__ __align__(16) __bf16 Ql[32*32];
  __shared__ __align__(16) float  Ps[32*SP];
  __shared__ __align__(16) __bf16 Pb[32*SP];
  int tid = threadIdx.x, w = tid >> 5, l = tid & 31;
  int b = blockIdx.x >> 3, h = blockIdx.x & 7;
  const float* base = qkv + (size_t)b * SS * 384;
  for (int i = tid; i < SP*16; i += 256) {
    int s = i >> 4, d = i & 15;
    float kv = 0.f, vv = 0.f;
    if (s < SS) {
      kv = base[(size_t)s*384 + 128 + h*16 + d];
      vv = base[(size_t)s*384 + 256 + h*16 + d];
    }
    Kl[i] = (__bf16)kv; Vl[i] = (__bf16)vv;
  }
  __syncthreads();
  for (int qc = 0; qc < 7; ++qc) {
    int qb = qc * 32;
    for (int i = tid; i < 32*32; i += 256) {
      int r = i >> 5, k = i & 31;
      int s = qb + r;
      float qv = (k < 16 && s < SS) ? base[(size_t)s*384 + h*16 + k] * 0.25f : 0.f;
      Ql[i] = (__bf16)qv;
    }
    __syncthreads();
    for (int tt = w; tt < 26; tt += 8) {       // scores = (Q*scale) @ K^T
      int mt = tt & 1, nt = tt >> 1;
      v16bf af  = load_a(&Ql[mt*16*32], 32);
      v16bf bfK = load_b_nk(&Kl[(nt*16)*16], 16, 16);
      v8f sc = wmma_bf16(af, bfK, zero8());
      int rowoff = (l < 16) ? 0 : 8;
      int n = nt*16 + (l & 15);
#pragma unroll
      for (int r = 0; r < 8; ++r) Ps[(mt*16 + rowoff + r)*SP + n] = sc[r];
    }
    __syncthreads();
    for (int i = tid; i < 32*8; i += 256)      // mask pad keys 200..207
      Ps[(i>>3)*SP + 200 + (i & 7)] = -3.0e38f;
    __syncthreads();
    { // softmax: wave w -> rows w*4..w*4+3, 8 lanes per row
      int r = w*4 + (l >> 3), li = l & 7;
      float mx = -3.0e38f;
      for (int c = li; c < SP; c += 8) mx = fmaxf(mx, Ps[r*SP + c]);
      mx = fmaxf(mx, __shfl_xor(mx, 1, 8));
      mx = fmaxf(mx, __shfl_xor(mx, 2, 8));
      mx = fmaxf(mx, __shfl_xor(mx, 4, 8));
      float sm = 0.f;
      for (int c = li; c < SP; c += 8) {
        float e = __expf(Ps[r*SP + c] - mx);
        Ps[r*SP + c] = e; sm += e;
      }
      sm += __shfl_xor(sm, 1, 8);
      sm += __shfl_xor(sm, 2, 8);
      sm += __shfl_xor(sm, 4, 8);
      float ri = 1.f / sm;
      for (int c = li; c < SP; c += 8) Pb[r*SP + c] = (__bf16)(Ps[r*SP + c] * ri);
    }
    __syncthreads();
    if (w < 2) { // PV: O[32,16] = P[32,208] @ V[208,16]
      v8f acc = zero8();
      for (int ks = 0; ks < 7; ++ks) {
        int kmax = SP - ks*32; if (kmax > 32) kmax = 32;
        v16bf af  = load_a_clamp(&Pb[w*16*SP + ks*32], SP, kmax);
        v16bf bfV = load_b_kn(&Vl[(ks*32)*16], 16, kmax);
        acc = wmma_bf16(af, bfV, acc);
      }
      int rowoff = (l < 16) ? 0 : 8;
      int d = l & 15;
#pragma unroll
      for (int r = 0; r < 8; ++r) {
        int s = qb + w*16 + rowoff + r;
        if (s < SS) Ob[((size_t)b*SS + s)*DDIM + h*16 + d] = (__bf16)acc[r];
      }
    }
    __syncthreads();
  }
}

// Residual add + LayerNorm (D=128): one wave per row, 8 rows/block.
__global__ __launch_bounds__(256) void k_add_ln(
    float* __restrict__ x, const float* __restrict__ y,
    const float* __restrict__ g, const float* __restrict__ bt,
    __bf16* __restrict__ xb)
{
  int w = threadIdx.x >> 5, l = threadIdx.x & 31;
  size_t row = (size_t)blockIdx.x * 8 + w;
  float4 xv = *(const float4*)&x[row*128 + l*4];
  float4 yv = *(const float4*)&y[row*128 + l*4];
  float v[4] = {xv.x+yv.x, xv.y+yv.y, xv.z+yv.z, xv.w+yv.w};
  float s = v[0]+v[1]+v[2]+v[3];
  for (int m = 1; m < 32; m <<= 1) s += __shfl_xor(s, m, 32);
  float mean = s * (1.f/128.f);
  float q = 0.f;
#pragma unroll
  for (int i = 0; i < 4; ++i) { float d = v[i]-mean; q += d*d; }
  for (int m = 1; m < 32; m <<= 1) q += __shfl_xor(q, m, 32);
  float rstd = rsqrtf(q * (1.f/128.f) + 1e-5f);
#pragma unroll
  for (int i = 0; i < 4; ++i) {
    int c = l*4 + i;
    float o = (v[i]-mean)*rstd*g[c] + bt[c];
    x[row*128 + c]  = o;
    xb[row*128 + c] = (__bf16)o;
  }
}

// Fused FFN: o = relu(x @ W1^T + b1) @ W2^T + b2 ; LDS-resident intermediate.
// All staging is bf16 b128 async copies.
__global__ __launch_bounds__(256) void k_ffn(
    const __bf16* __restrict__ xb, const __bf16* __restrict__ W1,
    const float* __restrict__ b1, const __bf16* __restrict__ W2,
    const float* __restrict__ b2, float* __restrict__ o)
{
  __shared__ __align__(16) __bf16 Al[64*128];
  __shared__ __align__(16) __bf16 Ws[64*32];
  __shared__ __align__(16) __bf16 Fl[64*64];
  __shared__ __align__(16) __bf16 W2s[128*32];
  int tid = threadIdx.x, w = tid >> 5, l = tid & 31;
  size_t m0 = (size_t)blockIdx.x * 64;
  for (int i = tid; i < 1024; i += 256)
    cp_b128(xb + m0*128 + i*8, &Al[i*8]);
  int rowoff = (l < 16) ? 0 : 8;
  v8f oacc[4];
#pragma unroll
  for (int nt = 0; nt < 4; ++nt) oacc[nt] = zero8();
  int srow = tid >> 2, skq = (tid & 3) * 8;
  for (int dc = 0; dc < 32; ++dc) {
    v8f facc[2]; facc[0] = zero8(); facc[1] = zero8();
    for (int ks = 0; ks < 4; ++ks) {
      cp_b128(W1 + (size_t)(dc*64 + srow)*128 + ks*32 + skq, &Ws[srow*32 + skq]);
      cp_wait();
      __syncthreads();
#pragma unroll
      for (int t = 0; t < 2; ++t) {
        int tt = w*2 + t, fmt = tt >> 2, fnt = tt & 3;
        v16bf af = load_a(&Al[fmt*16*128 + ks*32], 128);
        v16bf bw = load_b_nk(&Ws[(fnt*16)*32], 32, 32);
        facc[t] = wmma_bf16(af, bw, facc[t]);
      }
      __syncthreads();
    }
#pragma unroll
    for (int t = 0; t < 2; ++t) {     // bias + relu -> Fl (bf16)
      int tt = w*2 + t, fmt = tt >> 2, fnt = tt & 3;
      int c = fnt*16 + (l & 15);
      float bb = b1[dc*64 + c];
#pragma unroll
      for (int r = 0; r < 8; ++r) {
        float fv = fmaxf(facc[t][r] + bb, 0.f);
        Fl[(fmt*16 + rowoff + r)*64 + c] = (__bf16)fv;
      }
    }
    __syncthreads();
    for (int ks2 = 0; ks2 < 2; ++ks2) {
      int n2 = tid >> 1, kq2 = (tid & 1) * 16;
      cp_b128(W2 + (size_t)n2*2048 + dc*64 + ks2*32 + kq2,     &W2s[n2*32 + kq2]);
      cp_b128(W2 + (size_t)n2*2048 + dc*64 + ks2*32 + kq2 + 8, &W2s[n2*32 + kq2 + 8]);
      cp_wait();
      __syncthreads();
      v16bf af = load_a(&Fl[(w>>1)*16*64 + ks2*32], 64);
#pragma unroll
      for (int nt = 0; nt < 4; ++nt) {
        v16bf bw = load_b_nk(&W2s[((w&1)*64 + nt*16)*32], 32, 32);
        oacc[nt] = wmma_bf16(af, bw, oacc[nt]);
      }
      __syncthreads();
    }
  }
#pragma unroll
  for (int nt = 0; nt < 4; ++nt) {
    int c = (w&1)*64 + nt*16 + (l & 15);
    float bb = b2[c];
#pragma unroll
    for (int r = 0; r < 8; ++r) {
      size_t m = m0 + (w>>1)*16 + rowoff + r;
      o[m*128 + c] = oacc[nt][r] + bb;
    }
  }
}

// Mean pool over seq + concat: h[b, 0:128]=mean(xt), h[b,128:256]=mean(xi)
__global__ __launch_bounds__(256) void k_pool(
    const float* __restrict__ xt, const float* __restrict__ xi,
    float* __restrict__ hc, __bf16* __restrict__ hcb)
{
  int b = blockIdx.x, t = threadIdx.x;
  const float* src = (t < 128) ? xt : xi;
  int d = t & 127;
  float s = 0.f;
  for (int q = 0; q < SS; ++q) s += src[((size_t)b*SS + q)*128 + d];
  s *= (1.f/200.f);
  hc[b*256 + t]  = s;
  hcb[b*256 + t] = (__bf16)s;
}

// Final projection: out[(b*8+t)*V + n] = h[b]·W[n,:512] + bias[n] + t*W[n,512]
// W streamed once from HBM (non-temporal + prefetch); output stored non-temporal.
__global__ __launch_bounds__(256) void k_proj(
    const __bf16* __restrict__ hb, const float* __restrict__ W,
    const float* __restrict__ bias, float* __restrict__ out, int V)
{
  __shared__ __align__(16) __bf16 Hl[128*32];
  __shared__ __align__(16) __bf16 Wl[16*32];
  int tid = threadIdx.x, w = tid >> 5, l = tid & 31;
  int n0 = blockIdx.x * 16;
  v8f acc = zero8();
  int hrow = tid >> 1, hk = (tid & 1) * 16;
  for (int ks = 0; ks < 16; ++ks) {
    cp_b128(hb + hrow*512 + ks*32 + hk,     &Hl[hrow*32 + hk]);
    cp_b128(hb + hrow*512 + ks*32 + hk + 8, &Hl[hrow*32 + hk + 8]);
    if (tid < 64) {
      int n = tid >> 2, kq = (tid & 3) * 8;
      int wr = n0 + n; if (wr > V-1) wr = V-1;
      const float* wg = W + (size_t)wr*513 + ks*32 + kq;
      __builtin_prefetch(wg + 16*513, 0, 1);      // next block's rows
      __bf16* wd = &Wl[n*32 + kq];
#pragma unroll
      for (int i = 0; i < 8; ++i) wd[i] = (__bf16)__builtin_nontemporal_load(wg + i);
    }
    cp_wait();
    __syncthreads();
    v16bf af = load_a(&Hl[w*16*32], 32);
    v16bf bw = load_b_nk(Wl, 32, 32);
    acc = wmma_bf16(af, bw, acc);
    __syncthreads();
  }
  int n = n0 + (l & 15);
  if (n < V) {
    float bv = bias[n];
    float sl = W[(size_t)n*513 + 512];
    int rowoff = (l < 16) ? 0 : 8;
#pragma unroll
    for (int r = 0; r < 8; ++r) {
      int m = w*16 + rowoff + r;   // batch index
      float bval = acc[r] + bv;
#pragma unroll
      for (int t = 0; t < 8; ++t)
        __builtin_nontemporal_store(bval + (float)t * sl,
                                    &out[((size_t)m*8 + t)*V + n]);
    }
  }
}

// ---------------- host ----------------
extern "C" void kernel_launch(void* const* d_in, const int* in_sizes, int n_in,
                              void* d_out, int out_size, void* d_ws, size_t ws_size,
                              hipStream_t stream)
{
  (void)in_sizes; (void)n_in; (void)out_size; (void)ws_size;
  const int*   tseq = (const int*)d_in[0];
  const int*   iseq = (const int*)d_in[1];
  const float* te   = (const float*)d_in[2];
  const float* ie   = (const float*)d_in[3];
  const float* Wqkv = (const float*)d_in[4];
  const float* bqkv = (const float*)d_in[5];
  const float* Wo   = (const float*)d_in[6];
  const float* bo   = (const float*)d_in[7];
  const float* g1   = (const float*)d_in[8];
  const float* be1  = (const float*)d_in[9];
  const float* W1   = (const float*)d_in[10];
  const float* b1   = (const float*)d_in[11];
  const float* W2   = (const float*)d_in[12];
  const float* b2   = (const float*)d_in[13];
  const float* g2   = (const float*)d_in[14];
  const float* be2  = (const float*)d_in[15];
  const float* Wlin = (const float*)d_in[16];
  const float* blin = (const float*)d_in[17];
  const float* Wtab = (const float*)d_in[18];
  const float* btab = (const float*)d_in[19];
  const float* Widx = (const float*)d_in[20];
  const float* bidx = (const float*)d_in[21];
  float* out = (float*)d_out;

  char* p = (char*)d_ws;
  auto carve = [&](size_t bytes) { char* q = p; p += (bytes + 255) & ~(size_t)255; return q; };
  const size_t M = MRows;
  float*  xt  = (float*) carve(M*128*4);
  float*  xi  = (float*) carve(M*128*4);
  __bf16* xtb = (__bf16*)carve(M*128*2);
  __bf16* xib = (__bf16*)carve(M*128*2);
  float*  qkv = (float*) carve(M*384*4);
  __bf16* ab  = (__bf16*)carve(M*128*2);
  float*  osc = (float*) carve(M*128*4);
  float*  hc  = (float*) carve(128*256*4);
  __bf16* hcb = (__bf16*)carve(128*256*2);
  float*  h   = (float*) carve(128*512*4);
  __bf16* hb  = (__bf16*)carve(128*512*2);
  // bf16 weight mirrors (encoder + Wlin) so GEMM staging is pure b128 copies
  __bf16* Wqkvb = (__bf16*)carve((size_t)4*384*128*2);
  __bf16* Wob   = (__bf16*)carve((size_t)4*128*128*2);
  __bf16* W1b   = (__bf16*)carve((size_t)4*2048*128*2);
  __bf16* W2b   = (__bf16*)carve((size_t)4*128*2048*2);
  __bf16* Wlinb = (__bf16*)carve((size_t)512*256*2);

  k_cvt<<<512, 256, 0, stream>>>(Wqkv, Wqkvb, 4*384*128);
  k_cvt<<<256, 256, 0, stream>>>(Wo,   Wob,   4*128*128);
  k_cvt<<<1024, 256, 0, stream>>>(W1,  W1b,   4*2048*128);
  k_cvt<<<1024, 256, 0, stream>>>(W2,  W2b,   4*128*2048);
  k_cvt<<<256, 256, 0, stream>>>(Wlin, Wlinb, 512*256);

  k_gather<<<(unsigned)M, 256, 0, stream>>>(tseq, iseq, te, ie, xt, xi, xtb, xib);

  for (int e = 0; e < 2; ++e) {
    float*  x  = e ? xi  : xt;
    __bf16* xb = e ? xib : xtb;
    for (int l = 0; l < 2; ++l) {
      int off = e*2 + l;
      k_gemm<<<dim3(384/64, (unsigned)(M/64)), 256, 0, stream>>>(
          xb, Wqkvb + (size_t)off*384*128, bqkv + off*384, qkv,
          (__bf16*)nullptr, (int)M, 384, 128, 0);
      k_attn<<<128*8, 256, 0, stream>>>(qkv, ab);
      k_gemm<<<dim3(128/64, (unsigned)(M/64)), 256, 0, stream>>>(
          ab, Wob + (size_t)off*128*128, bo + off*128, osc,
          (__bf16*)nullptr, (int)M, 128, 128, 0);
      k_add_ln<<<(unsigned)(M/8), 256, 0, stream>>>(x, osc, g1 + off*128, be1 + off*128, xb);
      k_ffn<<<(unsigned)(M/64), 256, 0, stream>>>(
          xb, W1b + (size_t)off*2048*128, b1 + off*2048,
          W2b + (size_t)off*128*2048, b2 + off*128, osc);
      k_add_ln<<<(unsigned)(M/8), 256, 0, stream>>>(x, osc, g2 + off*128, be2 + off*128, xb);
    }
  }

  k_pool<<<128, 256, 0, stream>>>(xt, xi, hc, hcb);
  k_gemm<<<dim3(512/64, 128/64), 256, 0, stream>>>(hcb, Wlinb, blin, h, hb, 128, 512, 256, 1);
  k_proj<<<(1000 + 15)/16, 256, 0, stream>>>(hb, Wtab, btab, out, 1000);
  k_proj<<<(100000 + 15)/16, 256, 0, stream>>>(hb, Widx, bidx, out + (size_t)1024*1000, 100000);
}
